// SelfAttention_55121610277427
// MI455X (gfx1250) — matrix operand order repaired and checked by reading, not compiled, not run on hardware
//
#include <hip/hip_runtime.h>

#define B_DIM  8
#define C_DIM  256
#define N_DIM  4096
#define DQK    32
#define BN_EPS 1e-5f

typedef __attribute__((ext_vector_type(16))) _Float16 v16h;
typedef __attribute__((ext_vector_type(8)))  float    v8f;
typedef unsigned int u32x4 __attribute__((ext_vector_type(4)));
typedef int          i32x4 __attribute__((ext_vector_type(4)));
typedef int          i32x8 __attribute__((ext_vector_type(8)));

__device__ __forceinline__ v8f wmma_f16(v16h a, v16h b, v8f c) {
  // D = A(16x32 f16) x B(32x16 f16) + C(16x16 f32)
  return __builtin_amdgcn_wmma_f32_16x16x32_f16(false, a, false, b, (short)0, c, false, false);
}

// A-fragment (16x32 f16). Source row-major [row][k], stride_h halfs per row.
// Lane 0-15: M=lane, K = 0..15.  Lane 16-31: M=lane-16, K = 16..31.
__device__ __forceinline__ v16h load_a_frag(const _Float16* base, int stride_h, int lane) {
  const _Float16* p = base + (lane & 15) * stride_h + ((lane < 16) ? 0 : 16);
  v16h a;
#pragma unroll
  for (int t = 0; t < 16; ++t) a[t] = p[t];
  return a;
}

// B-fragment (32x16 f16). Source stored [n][k] (k contiguous), stride_h halfs per n-row.
__device__ __forceinline__ v16h load_b_frag(const _Float16* base, int stride_h, int lane) {
  const _Float16* p = base + (lane & 15) * stride_h + ((lane < 16) ? 0 : 16);
  v16h b;
#pragma unroll
  for (int t = 0; t < 16; ++t) b[t] = p[t];
  return b;
}

// Same, but from an f32 weight matrix (converted on the fly).
__device__ __forceinline__ v16h load_b_frag_f32(const float* base, int stride_f, int lane) {
  const float* p = base + (lane & 15) * stride_f + ((lane < 16) ? 0 : 16);
  v16h b;
#pragma unroll
  for (int t = 0; t < 16; ++t) b[t] = (_Float16)p[t];
  return b;
}

// ---------------------------------------------------------------------------
// TDM: 2-D tile load, global (f16 elements) -> LDS.  Descriptor per
// CDNA5 ISA.  data_size = 2 bytes.  Groups 2/3 zero
// (tile_dim2 = 0 => 2-D).  Tracked by TENSORcnt, in-order per wave.
// ---------------------------------------------------------------------------
__device__ __forceinline__ void tdm_load_2d_f16(
    unsigned int lds_addr, const void* gaddr,
    unsigned int tensor_d0, unsigned int tensor_d1,
    unsigned int tile_d0, unsigned int tile_d1, unsigned int stride_d0)
{
  const unsigned long long ga = (unsigned long long)(uintptr_t)gaddr;
  u32x4 g0;
  g0[0] = 1u;                                            // count=1
  g0[1] = lds_addr;                                      // lds_addr [63:32]
  g0[2] = (unsigned int)(ga & 0xFFFFFFFFu);              // global_addr [95:64]
  g0[3] = (unsigned int)((ga >> 32) & 0x1FFFFFFu) | (2u << 30); // [120:96] + type=2
  i32x8 g1;
  g1[0] = (int)(1u << 16);                               // data_size=1 (2 bytes)
  g1[1] = (int)((tensor_d0 & 0xFFFFu) << 16);            // tensor_dim0[15:0]
  g1[2] = (int)(((tensor_d0 >> 16) & 0xFFFFu) | ((tensor_d1 & 0xFFFFu) << 16));
  g1[3] = (int)(((tensor_d1 >> 16) & 0xFFFFu) | ((tile_d0 & 0xFFFFu) << 16));
  g1[4] = (int)(tile_d1 & 0xFFFFu);                      // tile_dim1; tile_dim2=0
  g1[5] = (int)stride_d0;                                // dim0_stride[31:0]
  g1[6] = 0;                                             // stride hi / dim1_stride lo
  g1[7] = 0;
  const i32x4 z4 = {0, 0, 0, 0};
#if __clang_major__ >= 23
  const i32x8 z8 = {0, 0, 0, 0, 0, 0, 0, 0};
  __builtin_amdgcn_tensor_load_to_lds(g0, g1, z4, z4, z8, 0);
#else
  __builtin_amdgcn_tensor_load_to_lds(g0, g1, z4, z4, 0);
#endif
}

// ---------------------------------------------------------------------------
// Kernel 1: fused 1x1-conv + BN for Q, K, V.  Q/K -> [b][n][32] f16,
// V -> transposed [b][c][n] f16 (so P*V B-fragments are contiguous later).
// ---------------------------------------------------------------------------
#define PROJ_M 128
#define LDX    264   // 256 + 8 halfs pad

__global__ __launch_bounds__(256) void proj_qkv_kernel(
    const float* __restrict__ x,
    const float* __restrict__ wq, const float* __restrict__ qs, const float* __restrict__ qb,
    const float* __restrict__ qm, const float* __restrict__ qv,
    const float* __restrict__ wk, const float* __restrict__ ksc, const float* __restrict__ kbi,
    const float* __restrict__ kme, const float* __restrict__ kva,
    const float* __restrict__ wv, const float* __restrict__ vsc, const float* __restrict__ vbi,
    const float* __restrict__ vme, const float* __restrict__ vva,
    _Float16* __restrict__ Qh, _Float16* __restrict__ Kh, _Float16* __restrict__ Vt)
{
  __shared__ _Float16 Xs[PROJ_M * LDX];

  const int tid = threadIdx.x;
  const int b  = blockIdx.y;
  const int n0 = blockIdx.x * PROJ_M;

  // Stage x[b, :, n0..n0+127] into Xs[m][c] (f16), coalesced float4 along n.
  for (int e = tid; e < (PROJ_M / 4) * C_DIM; e += 256) {
    const int c  = e >> 5;   // e / 32
    const int m4 = e & 31;
    const float4 xv = *(const float4*)(x + ((size_t)b * C_DIM + c) * N_DIM + n0 + m4 * 4);
    const int m = m4 * 4;
    Xs[(m + 0) * LDX + c] = (_Float16)xv.x;
    Xs[(m + 1) * LDX + c] = (_Float16)xv.y;
    Xs[(m + 2) * LDX + c] = (_Float16)xv.z;
    Xs[(m + 3) * LDX + c] = (_Float16)xv.w;
  }
  __syncthreads();

  const int wave = tid >> 5;
  const int lane = tid & 31;
  const int m0   = wave * 16;

  // 20 output tiles of 16: [Q0 Q1 | K0 K1 | V0..V15]
  for (int ot = 0; ot < 20; ++ot) {
    const float *w, *sc, *bi, *me, *va; int obase, which;
    if (ot < 2)      { w = wq; sc = qs;  bi = qb;  me = qm;  va = qv;  obase = ot * 16;       which = 0; }
    else if (ot < 4) { w = wk; sc = ksc; bi = kbi; me = kme; va = kva; obase = (ot - 2) * 16; which = 1; }
    else             { w = wv; sc = vsc; bi = vbi; me = vme; va = vva; obase = (ot - 4) * 16; which = 2; }

    v8f acc = {0.f, 0.f, 0.f, 0.f, 0.f, 0.f, 0.f, 0.f};
#pragma unroll 4
    for (int k0 = 0; k0 < C_DIM; k0 += 32) {
      const v16h a  = load_a_frag(Xs + m0 * LDX + k0, LDX, lane);
      const v16h bb = load_b_frag_f32(w + obase * C_DIM + k0, C_DIM, lane);
      acc = wmma_f16(a, bb, acc);
    }

    const int   ol   = obase + (lane & 15);
    const float inv  = sc[ol] * rsqrtf(va[ol] + BN_EPS);
    const float beta = bi[ol] - me[ol] * inv;
#pragma unroll
    for (int r = 0; r < 8; ++r) {
      const int   m   = m0 + r + ((lane < 16) ? 0 : 8);
      const int   n   = n0 + m;
      const float val = acc[r] * inv + beta;
      if (which == 0)      Qh[((size_t)b * N_DIM + n) * DQK + ol]   = (_Float16)val;
      else if (which == 1) Kh[((size_t)b * N_DIM + n) * DQK + ol]   = (_Float16)val;
      else                 Vt[((size_t)b * C_DIM + ol) * N_DIM + n] = (_Float16)val;
    }
  }
}

// ---------------------------------------------------------------------------
// Kernel 2: fused flash-attention.  Per WG: 64 query rows x 256 channels.
// 8 waves = 4 i-subtiles x 2 channel-halves.  K/V tiles streamed into
// double-buffered LDS by the Tensor Data Mover (wave 0 issues, TENSORcnt
// waits, workgroup barrier releases).  Online softmax, P*V via WMMA.
// ---------------------------------------------------------------------------
#define JT     32                 // j-tile size
#define NITER  (N_DIM / JT)

__global__ __launch_bounds__(256) void attn_kernel(
    const _Float16* __restrict__ Qh, const _Float16* __restrict__ Kh,
    const _Float16* __restrict__ Vt,
    const float* __restrict__ x, const float* __restrict__ gamma,
    float* __restrict__ out)
{
  __shared__ _Float16 KsS[2][JT * 32];      // K tile  [j_local][d]     (2 KB x2)
  __shared__ _Float16 VsS[2][C_DIM * JT];   // V tile  [c][j_local]     (16 KB x2)
  __shared__ _Float16 PsS[8][16 * JT];      // per-wave P tile [i][j_local]

  const int tid    = threadIdx.x;
  const int b      = blockIdx.y;
  const int i0     = blockIdx.x * 64;
  const int wave   = tid >> 5;
  const int lane   = tid & 31;
  const int i_base = i0 + (wave >> 1) * 16;
  const int c_base = (wave & 1) * 128;

  const _Float16* kbase = Kh + (size_t)b * N_DIM * DQK;
  const _Float16* vbase = Vt + (size_t)b * C_DIM * N_DIM;

  const v16h qfrag = load_a_frag(Qh + ((size_t)b * N_DIM + i_base) * DQK, DQK, lane);

  v8f   acc[8];
  float mrow[8], lsum[8];
#pragma unroll
  for (int t = 0; t < 8; ++t) {
    acc[t]  = (v8f){0.f, 0.f, 0.f, 0.f, 0.f, 0.f, 0.f, 0.f};
    mrow[t] = -1e30f;
    lsum[t] = 0.f;
  }
  const v8f vzero = {0.f, 0.f, 0.f, 0.f, 0.f, 0.f, 0.f, 0.f};

  // Prologue: TDM-load tile 0 into buffer 0 (wave 0 only; TDM ignores EXEC,
  // the uniform branch keeps other waves from issuing).
  if (wave == 0) {
    tdm_load_2d_f16((unsigned int)(uintptr_t)&KsS[0][0], kbase,
                    DQK, N_DIM, DQK, JT, DQK);
    tdm_load_2d_f16((unsigned int)(uintptr_t)&VsS[0][0], vbase,
                    N_DIM, C_DIM, JT, C_DIM, N_DIM);
  }

  for (int it = 0; it < NITER; ++it) {
    const int buf = it & 1;
    if (wave == 0) {
      if (it + 1 < NITER) {
        const int j1 = (it + 1) * JT;
        tdm_load_2d_f16((unsigned int)(uintptr_t)&KsS[buf ^ 1][0],
                        kbase + (size_t)j1 * DQK, DQK, N_DIM, DQK, JT, DQK);
        tdm_load_2d_f16((unsigned int)(uintptr_t)&VsS[buf ^ 1][0],
                        vbase + j1, N_DIM, C_DIM, JT, C_DIM, N_DIM);
        __builtin_amdgcn_s_wait_tensorcnt(2);  // tile `it` (oldest 2 ops) done
      } else {
        __builtin_amdgcn_s_wait_tensorcnt(0);
      }
    }
    __syncthreads();   // release all waves once tile `it` is in LDS

    // Scores: S[i, j..j+31] = Q(16x32) * K^T, two 16-col WMMAs (K depth = d).
    const v16h bk0 = load_b_frag(&KsS[buf][0],       32, lane);
    const v16h bk1 = load_b_frag(&KsS[buf][16 * 32], 32, lane);
    const v8f  s0  = wmma_f16(qfrag, bk0, vzero);
    const v8f  s1  = wmma_f16(qfrag, bk1, vzero);

    // Online softmax update (rows live in 16-lane halves of each acc VGPR).
    float alpha[8];
#pragma unroll
    for (int r = 0; r < 8; ++r) {
      float mx = fmaxf(s0[r], s1[r]);
      mx = fmaxf(mx, __shfl_xor(mx, 1));
      mx = fmaxf(mx, __shfl_xor(mx, 2));
      mx = fmaxf(mx, __shfl_xor(mx, 4));
      mx = fmaxf(mx, __shfl_xor(mx, 8));
      const float mn = fmaxf(mrow[r], mx);
      alpha[r] = __expf(mrow[r] - mn);
      mrow[r]  = mn;
      const float p0 = __expf(s0[r] - mn);
      const float p1 = __expf(s1[r] - mn);
      lsum[r] = lsum[r] * alpha[r] + p0 + p1;
      const int row = r + ((lane < 16) ? 0 : 8);
      const int col = lane & 15;
      PsS[wave][row * 32 + col]      = (_Float16)p0;
      PsS[wave][row * 32 + col + 16] = (_Float16)p1;
    }
#pragma unroll
    for (int ct = 0; ct < 8; ++ct)
#pragma unroll
      for (int r = 0; r < 8; ++r)
        acc[ct][r] *= alpha[r];

    // Same-wave LDS store -> load ordering for the P tile.
    asm volatile("s_wait_dscnt 0" ::: "memory");

    const v16h pfrag = load_a_frag(&PsS[wave][0], 32, lane);
#pragma unroll
    for (int ct = 0; ct < 8; ++ct) {
      const v16h bv = load_b_frag(&VsS[buf][(c_base + ct * 16) * 32], 32, lane);
      acc[ct] = wmma_f16(pfrag, bv, acc[ct]);
    }
    __syncthreads();   // all waves done with buf before tile it+2 overwrites it
  }

  // Finalize: normalize, fuse gamma*out + x, store [b][c][n].
  const float g = gamma[0];
  float rinv[8];
#pragma unroll
  for (int r = 0; r < 8; ++r) {
    float ls = lsum[r];
    ls += __shfl_xor(ls, 1);
    ls += __shfl_xor(ls, 2);
    ls += __shfl_xor(ls, 4);
    ls += __shfl_xor(ls, 8);
    rinv[r] = 1.0f / ls;
  }
#pragma unroll
  for (int ct = 0; ct < 8; ++ct) {
    const int c = c_base + ct * 16 + (lane & 15);
#pragma unroll
    for (int r = 0; r < 8; ++r) {
      const int    n   = i_base + r + ((lane < 16) ? 0 : 8);
      const size_t idx = ((size_t)b * C_DIM + c) * N_DIM + n;
      out[idx] = g * (acc[ct][r] * rinv[r]) + x[idx];
    }
  }
}

// ---------------------------------------------------------------------------
extern "C" void kernel_launch(void* const* d_in, const int* in_sizes, int n_in,
                              void* d_out, int out_size, void* d_ws, size_t ws_size,
                              hipStream_t stream) {
  (void)in_sizes; (void)n_in; (void)out_size; (void)ws_size;

  const float* x   = (const float*)d_in[0];
  const float* wq  = (const float*)d_in[1];
  const float* qs  = (const float*)d_in[2];
  const float* qb  = (const float*)d_in[3];
  const float* qm  = (const float*)d_in[4];
  const float* qv  = (const float*)d_in[5];
  const float* wk  = (const float*)d_in[6];
  const float* ksc = (const float*)d_in[7];
  const float* kbi = (const float*)d_in[8];
  const float* kme = (const float*)d_in[9];
  const float* kva = (const float*)d_in[10];
  const float* wv  = (const float*)d_in[11];
  const float* vsc = (const float*)d_in[12];
  const float* vbi = (const float*)d_in[13];
  const float* vme = (const float*)d_in[14];
  const float* vva = (const float*)d_in[15];
  const float* gmm = (const float*)d_in[16];

  // Workspace: Qh (2 MB) | Kh (2 MB) | Vt (16 MB), all f16.
  const size_t QK_BYTES = (size_t)B_DIM * N_DIM * DQK * sizeof(_Float16);
  char* ws = (char*)d_ws;
  _Float16* Qh = (_Float16*)ws;
  _Float16* Kh = (_Float16*)(ws + QK_BYTES);
  _Float16* Vt = (_Float16*)(ws + 2 * QK_BYTES);

  proj_qkv_kernel<<<dim3(N_DIM / PROJ_M, B_DIM), 256, 0, stream>>>(
      x, wq, qs, qb, qm, qv, wk, ksc, kbi, kme, kva, wv, vsc, vbi, vme, vva,
      Qh, Kh, Vt);

  attn_kernel<<<dim3(N_DIM / 64, B_DIM), 256, 0, stream>>>(
      Qh, Kh, Vt, x, gmm, (float*)d_out);
}
